// GatedEdgeInjection_77275051589820
// MI455X (gfx1250) — matrix operand order repaired and checked by reading, not compiled, not run on hardware
//
#include <hip/hip_runtime.h>

// ---------------------------------------------------------------------------
// Types for CDNA5 WMMA (wave32): bf16 A/B fragments (16 halves / lane),
// f32 C/D fragment (8 floats / lane).
// ---------------------------------------------------------------------------
typedef __attribute__((ext_vector_type(16))) __bf16 v16bf;
typedef __attribute__((ext_vector_type(8)))  float  v8f;
typedef __attribute__((ext_vector_type(4)))  unsigned int uv4;
typedef int gv4 __attribute__((vector_size(16)));   // builtin's int4 vector type
typedef unsigned short u16;

union ABf { uv4 q[2]; v16bf v; };

#if defined(__has_builtin)
#if __has_builtin(__builtin_amdgcn_global_load_async_to_lds_b128)
#define ASYNC_CP 1
#endif
#endif

// Async global->LDS 16B copy (gfx1250 GLOBAL_LOAD_ASYNC_TO_LDS_B128,
// tracked with ASYNCcnt). Signature (from hipcc diagnostic):
//   (gv4 addrspace(1)*, gv4 addrspace(3)*, imm offset, imm cpol)
// Fallback: plain load + ds_store (fenced by __syncthreads' dscnt wait).
__device__ __forceinline__ void cp_b128(const u16* gsrc, u16* ldst) {
#ifdef ASYNC_CP
  __builtin_amdgcn_global_load_async_to_lds_b128(
      (__attribute__((address_space(1))) gv4*)gsrc,
      (__attribute__((address_space(3))) gv4*)ldst, 0, 0);
#else
  *(uv4*)ldst = *(const uv4*)gsrc;
#endif
}

__device__ __forceinline__ void wait_async0() {
#ifdef ASYNC_CP
#if __has_builtin(__builtin_amdgcn_s_wait_asynccnt)
  __builtin_amdgcn_s_wait_asynccnt(0);
#else
  asm volatile("s_wait_asynccnt 0x0" ::: "memory");
#endif
#endif
}

__device__ __forceinline__ u16 f2bf(float f) {
  unsigned u = __float_as_uint(f);
  u += 0x7FFFu + ((u >> 16) & 1u);   // round to nearest even
  return (u16)(u >> 16);
}
__device__ __forceinline__ float bf2f(u16 h) {
  return __uint_as_float(((unsigned)h) << 16);
}

__device__ __forceinline__ v8f wmma_bf16(v16bf a, v16bf b, v8f c) {
  // D = A(16x32) * B(32x16) + C(16x16), f32 accumulate
  return __builtin_amdgcn_wmma_f32_16x16x32_bf16(
      /*neg_a=*/false, a, /*neg_b=*/false, b,
      /*c_mod=*/(short)0, c, /*reuse_a=*/false, /*reuse_b=*/false);
}

#define HWSZ 4096   // H*W

// ---------------------------------------------------------------------------
// NCHW f32 -> NHWC bf16 (LDS-tiled transpose; coalesced read and write)
// grid: 16 * 64 * 4 blocks (b, y, 64-channel slab), 256 threads
// ---------------------------------------------------------------------------
__global__ __launch_bounds__(256) void nchw_to_nhwc_bf16(
    const float* __restrict__ x, u16* __restrict__ xb) {
  __shared__ u16 s[64][65];
  const int t  = threadIdx.x;
  const int b  = blockIdx.x >> 8;
  const int y  = (blockIdx.x >> 2) & 63;
  const int c0 = (blockIdx.x & 3) << 6;
  {
    const int xcol = t & 63, cr = t >> 6;
#pragma unroll
    for (int it = 0; it < 16; ++it) {
      const int c = it * 4 + cr;
      s[c][xcol] = f2bf(x[(((b << 8) + c0 + c) << 12) + (y << 6) + xcol]);
    }
  }
  __syncthreads();
  {
    const int c = t & 63, xr = t >> 6;
#pragma unroll
    for (int it = 0; it < 16; ++it) {
      const int xcol = it * 4 + xr;
      xb[(((b << 12) + (y << 6) + xcol) << 8) + c0 + c] = s[c][xcol];
    }
  }
}

// ---------------------------------------------------------------------------
// Reorder weights [O, Cin, kh, kw] f32 -> bf16 in exact per-lane B-fragment
// order: flat index = (((s*ntiles + t)*32 + lane)*16 + half),
//   n = t*16 + (lane&15),  K = s*32 + (half>>3)*16 + (lane>>4)*8 + (half&7),
//   tap = K / Cin, c = K % Cin   (K-order: (kh,kw) major, channel minor)
// ---------------------------------------------------------------------------
__global__ __launch_bounds__(256) void reorder_w_kernel(
    const float* __restrict__ w, u16* __restrict__ wr,
    int Cin, int area, int ntiles, int total) {
  const int i = blockIdx.x * 256 + threadIdx.x;
  if (i >= total) return;
  const int half = i & 15;
  const int lane = (i >> 4) & 31;
  const int st   = i >> 9;
  const int t    = st % ntiles;
  const int s    = st / ntiles;
  const int n    = t * 16 + (lane & 15);
  const int K    = s * 32 + (half >> 3) * 16 + (lane >> 4) * 8 + (half & 7);
  const int tap  = K / Cin;
  const int c    = K - tap * Cin;
  wr[i] = f2bf(w[(n * Cin + c) * area + tap]);
}

// Fold conv-bias + BN into per-channel scale/shift
__global__ void prep_bn(const float* __restrict__ g, const float* __restrict__ bb,
                        const float* __restrict__ m, const float* __restrict__ v,
                        const float* __restrict__ cb,
                        float* __restrict__ scale, float* __restrict__ shift, int n) {
  const int i = blockIdx.x * 64 + threadIdx.x;
  if (i < n) {
    const float s = g[i] * rsqrtf(v[i] + 1e-5f);
    scale[i] = s;
    shift[i] = cb[i] * s + bb[i] - m[i] * s;
  }
}

// ---------------------------------------------------------------------------
// Implicit-GEMM conv3x3 + BN + ReLU, NHWC bf16 in/out, Cout = 64.
// Block = 256 threads = 8 waves; each wave: 16 consecutive spatial positions
// x all 64 output channels (4 WMMA N-tiles). K loop: 9 taps x (CIN/32)
// channel slices. The 4KB weight slice of each K-step is staged into LDS
// once per workgroup with async global->LDS copies (ASYNCcnt), double
// buffered so the DMA overlaps the WMMA math. grid = 512 blocks.
// ---------------------------------------------------------------------------
template <int CIN>
__global__ __launch_bounds__(256) void conv3x3_bn_relu_wmma(
    const u16* __restrict__ xin, const u16* __restrict__ wr,
    const float* __restrict__ scale, const float* __restrict__ shift,
    u16* __restrict__ out) {
  constexpr int CPT    = CIN >> 5;   // K-steps per tap
  constexpr int KSTEPS = 9 * CPT;
  __shared__ __align__(16) u16 wlds[2][2048];   // 2 x 4KB weight slices
  __shared__ __align__(16) u16 tile[8][1024];   // epilogue store staging

  const int tid  = threadIdx.x;
  const int lane = tid & 31, wave = tid >> 5;
  const int g = lane >> 4, l15 = lane & 15;
  const int base_p = blockIdx.x * 128 + wave * 16;
  const int p = base_p + l15;
  const int b = p >> 12, y = (p >> 6) & 63, xc = p & 63;
  v8f acc[4] = {};

  // Prologue: stage K-step 0 weights
  cp_b128(wr + tid * 8, &wlds[0][tid * 8]);
  wait_async0();
  __syncthreads();

  int s = 0;
  for (int kh = 0; kh < 3; ++kh) {
    const int iy = y + kh - 1;
    for (int kw = 0; kw < 3; ++kw) {
      const int ix = xc + kw - 1;
      const bool valid = ((unsigned)iy < 64u) && ((unsigned)ix < 64u);
      const int soff = (((b << 6) + iy) * 64 + ix) * CIN + g * 8;
#pragma unroll
      for (int cs = 0; cs < CPT; ++cs, ++s) {
        const int buf = s & 1;
        // Kick off async stage of the next weight slice
        if (s + 1 < KSTEPS)
          cp_b128(wr + (s + 1) * 2048 + tid * 8, &wlds[buf ^ 1][tid * 8]);
        // A fragment: 16x32 bf16; per lane two contiguous 8-half runs
        ABf a;
        if (valid) {
          const u16* sp = xin + soff + cs * 32;
          a.q[0] = *(const uv4*)sp;               // K = g*8 .. g*8+7
          a.q[1] = *(const uv4*)(sp + 16);        // K = 16+g*8 ..
        } else {
          uv4 z = {0u, 0u, 0u, 0u};
          a.q[0] = z; a.q[1] = z;
        }
        const uv4* wq = (const uv4*)wlds[buf];
        int bi = lane * 2;
#pragma unroll
        for (int t = 0; t < 4; ++t) {
          ABf bf;
          bf.q[0] = wq[bi]; bf.q[1] = wq[bi + 1];
          acc[t] = wmma_bf16(a.v, bf.v, acc[t]);
          bi += 64;
        }
        wait_async0();
        __syncthreads();
      }
    }
  }
  // Epilogue: BN+ReLU, restage through wave-private LDS so the NHWC store
  // (16 spatial * 64ch = contiguous 2KB) is four coalesced b128 stores.
#pragma unroll
  for (int t = 0; t < 4; ++t) {
    const int n = t * 16 + l15;
    const float sc = scale[n], sh = shift[n];
#pragma unroll
    for (int r = 0; r < 8; ++r) {
      float f = fmaf(acc[t][r], sc, sh);
      f = fmaxf(f, 0.f);
      tile[wave][(g * 8 + r) * 64 + n] = f2bf(f);
    }
  }
  uv4* dst = (uv4*)(out + base_p * 64);
  const uv4* srct = (const uv4*)tile[wave];
#pragma unroll
  for (int it = 0; it < 4; ++it) dst[it * 32 + lane] = srct[it * 32 + lane];
}

// ---------------------------------------------------------------------------
// Pool x (NCHW f32) over H,W: one block per (b,c)
// ---------------------------------------------------------------------------
__global__ __launch_bounds__(256) void pool_nchw(const float* __restrict__ x,
                                                 float* __restrict__ pool) {
  __shared__ float red[256];
  const long base = (long)blockIdx.x * HWSZ;
  float s = 0.f;
  for (int i = threadIdx.x; i < HWSZ; i += 256) s += x[base + i];
  red[threadIdx.x] = s;
  __syncthreads();
  for (int st = 128; st > 0; st >>= 1) {
    if (threadIdx.x < st) red[threadIdx.x] += red[threadIdx.x + st];
    __syncthreads();
  }
  if (threadIdx.x == 0) pool[blockIdx.x] = red[0] * (1.f / (float)HWSZ);
}

// Pool ef (NHWC bf16, 64 ch) over H,W: one block per (b,c), 1024 blocks
__global__ __launch_bounds__(256) void pool_nhwc_bf16(const u16* __restrict__ ef,
                                                      float* __restrict__ pool) {
  __shared__ float red[256];
  const int b = blockIdx.x >> 6, c = blockIdx.x & 63;
  const u16* p = ef + (long)b * HWSZ * 64 + c;
  float s = 0.f;
  for (int i = threadIdx.x; i < HWSZ; i += 256) s += bf2f(p[i * 64]);
  red[threadIdx.x] = s;
  __syncthreads();
  for (int st = 128; st > 0; st >>= 1) {
    if (threadIdx.x < st) red[threadIdx.x] += red[threadIdx.x + st];
    __syncthreads();
  }
  if (threadIdx.x == 0) pool[blockIdx.x] = red[0] * (1.f / (float)HWSZ);
}

// ---------------------------------------------------------------------------
// Gate MLP (tiny, exact f32): concat -> FC(320->128)+BN+ReLU -> FC(128->256)
// -> sigmoid. Single block of 256 threads.
// ---------------------------------------------------------------------------
__global__ __launch_bounds__(256) void gate_kernel(
    const float* __restrict__ xpool, const float* __restrict__ epool,
    const float* __restrict__ g1w, const float* __restrict__ g1b,
    const float* __restrict__ gbng, const float* __restrict__ gbnb,
    const float* __restrict__ gbnm, const float* __restrict__ gbnv,
    const float* __restrict__ g2w, const float* __restrict__ g2b,
    float* __restrict__ gate) {
  __shared__ float gsh[16 * 320];
  __shared__ float hsh[16 * 128];
  const int t = threadIdx.x;
  for (int i = t; i < 16 * 320; i += 256) {
    const int b = i / 320, j = i - b * 320;
    gsh[i] = (j < 256) ? xpool[b * 256 + j] : epool[b * 64 + (j - 256)];
  }
  __syncthreads();
  for (int i = t; i < 16 * 128; i += 256) {
    const int b = i >> 7, j = i & 127;
    float s = g1b[j];
    const float* wrow = g1w + j * 320;
    const float* gr = gsh + b * 320;
    for (int k = 0; k < 320; ++k) s = fmaf(gr[k], wrow[k], s);
    const float sc = gbng[j] * rsqrtf(gbnv[j] + 1e-5f);
    s = (s - gbnm[j]) * sc + gbnb[j];
    hsh[i] = fmaxf(s, 0.f);
  }
  __syncthreads();
  for (int i = t; i < 16 * 256; i += 256) {
    const int b = i >> 8, o = i & 255;
    float s = g2b[o];
    const float* wrow = g2w + o * 128;
    const float* hr = hsh + b * 128;
    for (int k = 0; k < 128; ++k) s = fmaf(hr[k], wrow[k], s);
    gate[i] = 1.f / (1.f + expf(-s));
  }
}

// ---------------------------------------------------------------------------
// Output: edge_enh = ef(NHWC bf16) @ out_w^T (K=64, N=256) via WMMA, then
// out(NCHW f32) = x + gate * (edge_enh + out_b). Each wave: 16 spatial x 256
// channels (16 WMMA tiles, 2 K-steps). Weights staged per K-step into LDS
// with async copies; epilogue transposes 16x16 tiles in wave-private LDS so
// NCHW reads/writes are spatially coalesced.
// ---------------------------------------------------------------------------
__global__ __launch_bounds__(256) void out_gemm_kernel(
    const u16* __restrict__ ef, const u16* __restrict__ wor,
    const float* __restrict__ xin, const float* __restrict__ gate,
    const float* __restrict__ outb, float* __restrict__ out) {
  __shared__ __align__(16) u16 wl[2][8192];   // 2 x 16KB weight slices
  __shared__ float ot[8][16][17];
  const int tid  = threadIdx.x;
  const int lane = tid & 31, wave = tid >> 5;
  const int g = lane >> 4, l15 = lane & 15;
  const int base_p = blockIdx.x * 128 + wave * 16;
  const int p = base_p + l15;
  const int b = base_p >> 12;
  const int prem = base_p & 4095;
  v8f acc[16] = {};

  // Prologue: stage K-step 0 weights (16KB = 4 x b128 per thread)
#pragma unroll
  for (int it = 0; it < 4; ++it)
    cp_b128(wor + (it * 256 + tid) * 8, &wl[0][(it * 256 + tid) * 8]);
  wait_async0();
  __syncthreads();

#pragma unroll
  for (int ks = 0; ks < 2; ++ks) {
    if (ks == 0) {
#pragma unroll
      for (int it = 0; it < 4; ++it)
        cp_b128(wor + 8192 + (it * 256 + tid) * 8, &wl[1][(it * 256 + tid) * 8]);
    }
    ABf a;
    const u16* sp = ef + p * 64 + ks * 32 + g * 8;
    a.q[0] = *(const uv4*)sp;
    a.q[1] = *(const uv4*)(sp + 16);
    const uv4* wq = (const uv4*)wl[ks];
#pragma unroll
    for (int t = 0; t < 16; ++t) {
      ABf bf;
      const int bi = (t * 32 + lane) * 2;
      bf.q[0] = wq[bi]; bf.q[1] = wq[bi + 1];
      acc[t] = wmma_bf16(a.v, bf.v, acc[t]);
    }
    wait_async0();
    __syncthreads();
  }
  for (int t = 0; t < 16; ++t) {
#pragma unroll
    for (int r = 0; r < 8; ++r) ot[wave][g * 8 + r][l15] = acc[t][r];
    const int n0 = t * 16;
#pragma unroll
    for (int it = 0; it < 8; ++it) {
      const int cl = it * 2 + g;
      const int n  = n0 + cl;
      const int gi = (b * 256 + n) * 4096 + prem + l15;
      const float e = ot[wave][l15][cl] + outb[n];
      out[gi] = fmaf(gate[b * 256 + n], e, xin[gi]);
    }
  }
}

// ---------------------------------------------------------------------------
// Host launcher
// ---------------------------------------------------------------------------
extern "C" void kernel_launch(void* const* d_in, const int* in_sizes, int n_in,
                              void* d_out, int out_size, void* d_ws, size_t ws_size,
                              hipStream_t stream) {
  (void)in_sizes; (void)n_in; (void)out_size; (void)ws_size;
  const float* x    = (const float*)d_in[0];
  const float* ec1w = (const float*)d_in[1];
  const float* ec1b = (const float*)d_in[2];
  const float* bn1g = (const float*)d_in[3];
  const float* bn1b = (const float*)d_in[4];
  const float* bn1m = (const float*)d_in[5];
  const float* bn1v = (const float*)d_in[6];
  const float* ec2w = (const float*)d_in[7];
  const float* ec2b = (const float*)d_in[8];
  const float* bn2g = (const float*)d_in[9];
  const float* bn2b = (const float*)d_in[10];
  const float* bn2m = (const float*)d_in[11];
  const float* bn2v = (const float*)d_in[12];
  const float* g1w  = (const float*)d_in[13];
  const float* g1b  = (const float*)d_in[14];
  const float* gbng = (const float*)d_in[15];
  const float* gbnb = (const float*)d_in[16];
  const float* gbnm = (const float*)d_in[17];
  const float* gbnv = (const float*)d_in[18];
  const float* g2w  = (const float*)d_in[19];
  const float* g2b  = (const float*)d_in[20];
  const float* outw = (const float*)d_in[21];
  const float* outbv= (const float*)d_in[22];
  float* out = (float*)d_out;

  // Workspace carve-up (all 256B aligned)
  char* ws = (char*)d_ws;
  u16* xb   = (u16*)ws;   ws += 33554432;  // x NHWC bf16 [16,4096,256]
  u16* ef1  = (u16*)ws;   ws += 8388608;   // edge feat 1 NHWC bf16 [16,4096,64]
  u16* ef2  = (u16*)ws;   ws += 8388608;   // edge feat 2 NHWC bf16
  u16* w1r  = (u16*)ws;   ws += 294912;    // conv1 weights, 72 ksteps x 4 ntiles
  u16* w2r  = (u16*)ws;   ws += 73728;     // conv2 weights, 18 ksteps x 4 ntiles
  u16* wor  = (u16*)ws;   ws += 32768;     // out 1x1 weights, 2 ksteps x 16 ntiles
  float* sc1   = (float*)ws; ws += 256;
  float* sh1   = (float*)ws; ws += 256;
  float* sc2   = (float*)ws; ws += 256;
  float* sh2   = (float*)ws; ws += 256;
  float* xpool = (float*)ws; ws += 16384;  // [16,256]
  float* epool = (float*)ws; ws += 4096;   // [16,64]
  float* gatep = (float*)ws; ws += 16384;  // [16,256]

  // 1) layout conversion + weight reorders + BN folds (independent)
  nchw_to_nhwc_bf16<<<4096, 256, 0, stream>>>(x, xb);
  reorder_w_kernel<<<576, 256, 0, stream>>>(ec1w, w1r, 256, 9, 4, 147456);
  reorder_w_kernel<<<144, 256, 0, stream>>>(ec2w, w2r, 64, 9, 4, 36864);
  reorder_w_kernel<<<64, 256, 0, stream>>>(outw, wor, 64, 1, 16, 16384);
  prep_bn<<<1, 64, 0, stream>>>(bn1g, bn1b, bn1m, bn1v, ec1b, sc1, sh1, 64);
  prep_bn<<<1, 64, 0, stream>>>(bn2g, bn2b, bn2m, bn2v, ec2b, sc2, sh2, 64);
  pool_nchw<<<4096, 256, 0, stream>>>(x, xpool);

  // 2) edge branch: conv -> BN -> ReLU, twice (WMMA bf16, async weight DMA)
  conv3x3_bn_relu_wmma<256><<<512, 256, 0, stream>>>(xb, w1r, sc1, sh1, ef1);
  conv3x3_bn_relu_wmma<64><<<512, 256, 0, stream>>>(ef1, w2r, sc2, sh2, ef2);

  // 3) gate branch
  pool_nhwc_bf16<<<1024, 256, 0, stream>>>(ef2, epool);
  gate_kernel<<<1, 256, 0, stream>>>(xpool, epool, g1w, g1b, gbng, gbnb,
                                     gbnm, gbnv, g2w, g2b, gatep);

  // 4) gated residual output (WMMA bf16 1x1 conv fused with add)
  out_gemm_kernel<<<512, 256, 0, stream>>>(ef2, wor, x, gatep, outbv, out);
}
